// MultiheadSelfAttentionDec_89524298317920
// MI455X (gfx1250) — matrix operand order repaired and checked by reading, not compiled, not run on hardware
//
#include <hip/hip_runtime.h>
#include <hip/hip_bf16.h>

// ---------------------------------------------------------------------------
// MI455X (gfx1250) fused multi-head self-attention.
//   - All matmuls via v_wmma_f32_16x16x32_f16 (wave32), f16 data / f32 accum.
//   - GEMM: 32x64 per wave (8 WMMA / K-step, B-frags reused across M).
//   - Flash attention: K tiles staged LDS via TDM tensor_load_to_lds with
//     double buffering (s_wait_tensorcnt), P restaged via LDS (s_wait_dscnt).
// ---------------------------------------------------------------------------

#define S_ 3072
#define C_ 2048
#define D_ 128
#define H_ 16
#define EPS_ 1.1920929e-07f

typedef __attribute__((ext_vector_type(16))) _Float16 v16h;
typedef __attribute__((ext_vector_type(8)))  float    v8f;
typedef __attribute__((ext_vector_type(4)))  int      int4v;
typedef __attribute__((ext_vector_type(8)))  int      int8v;

// --- Fragment loaders (layouts per CDNA5 ISA 7.12.2, wave32) ----------------
// A (16x32 f16, MxK): lane holds row M = lane&15.
//   elements 0..7  = K (lane<16 ? 0 : 8)  + j
//   elements 8..15 = K 16 + (lane<16?0:8) + j
__device__ __forceinline__ v16h load_a_frag(const _Float16* __restrict__ base,
                                            int ld, int row0, int k0) {
  const int lane = threadIdx.x & 31;
  const int hi = lane >> 4, lo = lane & 15;
  const _Float16* p = base + (size_t)(row0 + lo) * ld + k0 + hi * 8;
  union { v16h v; float4 f4[2]; } u;
  u.f4[0] = *(const float4*)p;
  u.f4[1] = *(const float4*)(p + 16);
  return u.v;
}

// B (32x16 f16, KxN): lane holds column N = lane&15,
//   elements 0..7  = K (lane<16 ? 0 : 8)  + j
//   elements 8..15 = K 16 + (lane<16?0:8) + j
__device__ __forceinline__ v16h load_b_frag(const _Float16* __restrict__ base,
                                            int ld, int n0, int k0) {
  const int lane = threadIdx.x & 31;
  const int hi = lane >> 4, lo = lane & 15;
  const _Float16* p = base + (size_t)(n0 + lo) * ld + k0 + hi * 8;
  union { v16h v; float4 f4[2]; } u;
  u.f4[0] = *(const float4*)p;
  u.f4[1] = *(const float4*)(p + 16);
  return u.v;
}

__device__ __forceinline__ v8f zero_v8f() {
  v8f z;
#pragma unroll
  for (int i = 0; i < 8; ++i) z[i] = 0.0f;
  return z;
}

// --- TDM: 2-D tile DMA global -> LDS (descriptor per ISA 08_async_tensor) ---
// data_size = 8B units. tile = rows x row_qwords; global row stride in qwords.
__device__ __forceinline__ void tdm_load_tile_2d(const void* gptr, void* lptr,
                                                 int rows, int row_qwords,
                                                 int row_stride_qwords,
                                                 int tensor_rows) {
  const unsigned long long ga = (unsigned long long)(uintptr_t)gptr;
  const unsigned la = (unsigned)(uintptr_t)lptr;
  const unsigned td0 = (unsigned)row_stride_qwords;  // dim0 bound covers row
  const unsigned td1 = (unsigned)tensor_rows;
  int4v g0;
  g0[0] = 1;                                   // count=1 (valid user D#)
  g0[1] = (int)la;                             // lds_addr
  g0[2] = (int)(unsigned)ga;                   // global_addr[31:0]
  g0[3] = (int)(((unsigned)(ga >> 32) & 0x01FFFFFFu) | 0x80000000u); // type=2
  int8v g1;
  g1[0] = 3 << 16;                             // data_size = 3 (8 bytes)
  g1[1] = (int)((td0 & 0xFFFFu) << 16);        // tensor_dim0[15:0]
  g1[2] = (int)((td0 >> 16) | ((td1 & 0xFFFFu) << 16)); // dim0 hi | dim1 lo
  g1[3] = (int)(((td1 >> 16) & 0xFFFFu) | ((unsigned)row_qwords << 16));
  g1[4] = rows & 0xFFFF;                       // tile_dim1 (tile_dim2 = 0)
  g1[5] = (int)td0;                            // tensor_dim0_stride[31:0]
  g1[6] = 0;
  g1[7] = 0;
  asm volatile("tensor_load_to_lds %0, %1" :: "s"(g0), "s"(g1) : "memory");
}

// --- f32 -> f16 convert -----------------------------------------------------
__global__ void f32_to_f16_kernel(const float* __restrict__ in,
                                  _Float16* __restrict__ out, int n) {
  int i = blockIdx.x * blockDim.x + threadIdx.x;
  if (i < n) out[i] = (_Float16)in[i];
}

// --- GEMM: Out[M,N] = A[M,K] @ W[N,K]^T + bias ------------------------------
// One wave computes a 32x64 tile (2x4 D-fragments, 8 WMMA per K-step with the
// 4 B-fragments reused across both M rows). 8 waves / block.
template <typename OutT>
__global__ void __launch_bounds__(256)
wmma_gemm_xwt(const _Float16* __restrict__ A, const _Float16* __restrict__ W,
              const float* __restrict__ bias, OutT* __restrict__ Out,
              int M, int N, int K) {
  const int wave = (blockIdx.x * blockDim.x + threadIdx.x) >> 5;
  const int nStrips = N >> 6;
  const int m0 = (wave / nStrips) << 5;   // 32 rows
  const int n0 = (wave % nStrips) << 6;   // 64 cols
  if (m0 >= M) return;
  const int lane = threadIdx.x & 31;
  const int hi = lane >> 4, lo = lane & 15;

  v8f acc[2][4];
#pragma unroll
  for (int r = 0; r < 2; ++r)
#pragma unroll
    for (int j = 0; j < 4; ++j) acc[r][j] = zero_v8f();

  for (int k0 = 0; k0 < K; k0 += 32) {
    v16h a0 = load_a_frag(A, K, m0, k0);
    v16h a1 = load_a_frag(A, K, m0 + 16, k0);
    if (k0 + 32 < K) {  // global_prefetch_b8 toward next K-slab
      __builtin_prefetch(A + (size_t)(m0 + lane) * K + k0 + 32, 0, 1);
      __builtin_prefetch(W + (size_t)(n0 + lane) * K + k0 + 32, 0, 1);
    }
#pragma unroll
    for (int j = 0; j < 4; ++j) {
      v16h b = load_b_frag(W, K, n0 + 16 * j, k0);
      acc[0][j] = __builtin_amdgcn_wmma_f32_16x16x32_f16(
          false, a0, false, b, (short)0, acc[0][j], false, false);
      acc[1][j] = __builtin_amdgcn_wmma_f32_16x16x32_f16(
          false, a1, false, b, (short)0, acc[1][j], false, false);
    }
  }
  // D layout: lane holds column n, VGPR v holds row m = v + 8*(lane>=16)
#pragma unroll
  for (int r = 0; r < 2; ++r)
#pragma unroll
    for (int j = 0; j < 4; ++j) {
      const int n = n0 + 16 * j + lo;
      const float bv = bias ? bias[n] : 0.0f;
#pragma unroll
      for (int v = 0; v < 8; ++v) {
        const int m = m0 + 16 * r + v + 8 * hi;
        Out[(size_t)m * N + n] = (OutT)(acc[r][j][v] + bv);
      }
    }
}

// --- per-(s,h) RMS-norm + RoPE, in place on f16 Q/K -------------------------
__global__ void __launch_bounds__(256)
norm_rope_kernel(_Float16* __restrict__ q, _Float16* __restrict__ k,
                 const float* __restrict__ rope,
                 const float* __restrict__ qw, const float* __restrict__ kw) {
  const int wave = (blockIdx.x * blockDim.x + threadIdx.x) >> 5;
  const int lane = threadIdx.x & 31;
  const int s = wave / H_;
  const int h = wave % H_;
  const int d0 = lane * 4;  // 32 lanes * 4 = 128 = D
  const float* rp = rope + (size_t)s * (D_ / 2) * 4;  // [i][a][b] -> i*4+a*2+b
  const float c0 = rp[(2 * lane) * 4 + 0];
  const float s0 = rp[(2 * lane) * 4 + 2];
  const float c1 = rp[(2 * lane + 1) * 4 + 0];
  const float s1 = rp[(2 * lane + 1) * 4 + 2];

#pragma unroll
  for (int which = 0; which < 2; ++which) {
    _Float16* row = (which ? k : q) + (size_t)s * C_ + h * D_;
    const float* w = which ? kw : qw;
    float x0 = (float)row[d0 + 0], x1 = (float)row[d0 + 1];
    float x2 = (float)row[d0 + 2], x3 = (float)row[d0 + 3];
    float ss = x0 * x0 + x1 * x1 + x2 * x2 + x3 * x3;
    ss += __shfl_xor(ss, 1);  ss += __shfl_xor(ss, 2);
    ss += __shfl_xor(ss, 4);  ss += __shfl_xor(ss, 8);
    ss += __shfl_xor(ss, 16);
    const float inv = rsqrtf(ss * (1.0f / D_) + EPS_);
    x0 *= inv * w[d0 + 0]; x1 *= inv * w[d0 + 1];
    x2 *= inv * w[d0 + 2]; x3 *= inv * w[d0 + 3];
    row[d0 + 0] = (_Float16)(c0 * x0 - s0 * x1);
    row[d0 + 1] = (_Float16)(s0 * x0 + c0 * x1);
    row[d0 + 2] = (_Float16)(c1 * x2 - s1 * x3);
    row[d0 + 3] = (_Float16)(s1 * x2 + c1 * x3);
  }
}

// --- V transpose: [S][H*D] -> per-head [H][D][S] (K-contiguous PV B-frags) --
__global__ void transpose_v_kernel(const _Float16* __restrict__ vraw,
                                   _Float16* __restrict__ vt) {
  int i = blockIdx.x * blockDim.x + threadIdx.x;
  if (i >= S_ * C_) return;
  const int s = i / C_, c = i % C_;
  vt[(size_t)c * S_ + s] = vraw[i];
}

// --- flash attention: one wave per (head, 16-query tile); 2 waves/block -----
// K tiles (32x128 f16) double-buffered into LDS by the Tensor Data Mover.
__global__ void __launch_bounds__(64)
flash_attn_kernel(const _Float16* __restrict__ qh, const _Float16* __restrict__ kh,
                  const _Float16* __restrict__ vt, _Float16* __restrict__ oh) {
  __shared__ __align__(16) _Float16 ktile[2][2][32 * D_];  // [wave][buf] 8KB
  __shared__ __align__(16) _Float16 ptile[2][16 * 32];     // [wave] P tile
  const int wid = threadIdx.x >> 5;
  const int lane = threadIdx.x & 31;
  const int hi = lane >> 4, lo = lane & 15;
  const int wave = blockIdx.x * 2 + wid;
  const int h = wave / (S_ / 16);
  const int m0 = (wave % (S_ / 16)) << 4;

  const _Float16* qbase = qh + h * D_;
  const _Float16* kbase = kh + h * D_;
  const _Float16* vbase = vt + (size_t)h * D_ * S_;
  _Float16* p_lds = ptile[wid];

  v16h a_q[4];
#pragma unroll
  for (int c = 0; c < 4; ++c) a_q[c] = load_a_frag(qbase, C_, m0, 32 * c);

  v8f acc[8];
#pragma unroll
  for (int n = 0; n < 8; ++n) acc[n] = zero_v8f();
  float row_m[8], row_s[8];
#pragma unroll
  for (int v = 0; v < 8; ++v) { row_m[v] = -INFINITY; row_s[v] = 0.0f; }

  const float scale = 0.08838834764831845f;  // 1/sqrt(128)
  const int NT = S_ / 32;

  // TDM-prefetch first K tile (32 keys x 128 d; row stride C_*2 bytes)
  tdm_load_tile_2d(kbase, ktile[wid][0], 32, D_ / 4, C_ / 4, S_);
  asm volatile("s_wait_tensorcnt 0x0" ::: "memory");

  for (int kt = 0; kt < NT; ++kt) {
    const int t0 = kt * 32;
    const _Float16* kcur = ktile[wid][kt & 1];
    if (kt + 1 < NT)  // DMA next K tile into the other buffer while computing
      tdm_load_tile_2d(kbase + (size_t)(t0 + 32) * C_, ktile[wid][(kt + 1) & 1],
                       32, D_ / 4, C_ / 4, S_);

    // scores 16x32 = two 16x16 D fragments; B-frags from LDS K tile (ld=128)
    v8f sf[2];
#pragma unroll
    for (int j = 0; j < 2; ++j) {
      sf[j] = zero_v8f();
#pragma unroll
      for (int c = 0; c < 4; ++c) {
        v16h b = load_b_frag(kcur, D_, 16 * j, 32 * c);
        sf[j] = __builtin_amdgcn_wmma_f32_16x16x32_f16(
            false, a_q[c], false, b, (short)0, sf[j], false, false);
      }
    }
    // online softmax; D-fragment rows m = v + 8*hi, cols striped over lanes&15
    float alpha[8];
#pragma unroll
    for (int v = 0; v < 8; ++v) {
      float s0 = sf[0][v] * scale, s1 = sf[1][v] * scale;
      float tmax = fmaxf(s0, s1);
      tmax = fmaxf(tmax, __shfl_xor(tmax, 1));
      tmax = fmaxf(tmax, __shfl_xor(tmax, 2));
      tmax = fmaxf(tmax, __shfl_xor(tmax, 4));
      tmax = fmaxf(tmax, __shfl_xor(tmax, 8));
      const float mnew = fmaxf(row_m[v], tmax);
      alpha[v] = __expf(row_m[v] - mnew);
      const float p0 = __expf(s0 - mnew);
      const float p1 = __expf(s1 - mnew);
      float ps = p0 + p1;
      ps += __shfl_xor(ps, 1);  ps += __shfl_xor(ps, 2);
      ps += __shfl_xor(ps, 4);  ps += __shfl_xor(ps, 8);
      row_s[v] = row_s[v] * alpha[v] + ps;
      row_m[v] = mnew;
      const int row = v + 8 * hi;
      p_lds[row * 32 + lo]      = (_Float16)p0;   // row-major 16x32 P tile
      p_lds[row * 32 + 16 + lo] = (_Float16)p1;
    }
    asm volatile("s_wait_dscnt 0x0" ::: "memory");  // P stores -> A-frag loads
    v16h a_p = load_a_frag(p_lds, 32, 0, 0);
    // out += P @ V  (Vt gives K(=t)-contiguous B fragments)
#pragma unroll
    for (int n = 0; n < 8; ++n) {
#pragma unroll
      for (int v = 0; v < 8; ++v) acc[n][v] *= alpha[v];
      v16h b = load_b_frag(vbase, S_, 16 * n, t0);
      acc[n] = __builtin_amdgcn_wmma_f32_16x16x32_f16(
          false, a_p, false, b, (short)0, acc[n], false, false);
    }
    if (kt + 1 < NT)  // next tile's DMA must be done before we read it
      asm volatile("s_wait_tensorcnt 0x0" ::: "memory");
  }
#pragma unroll
  for (int v = 0; v < 8; ++v) row_s[v] = 1.0f / row_s[v];
#pragma unroll
  for (int n = 0; n < 8; ++n)
#pragma unroll
    for (int v = 0; v < 8; ++v) {
      const int m = m0 + v + 8 * hi;
      oh[(size_t)m * C_ + h * D_ + 16 * n + lo] = (_Float16)(acc[n][v] * row_s[v]);
    }
}

// ---------------------------------------------------------------------------
extern "C" void kernel_launch(void* const* d_in, const int* in_sizes, int n_in,
                              void* d_out, int out_size, void* d_ws, size_t ws_size,
                              hipStream_t stream) {
  (void)in_sizes; (void)n_in; (void)out_size; (void)ws_size;
  const float* x    = (const float*)d_in[0];
  const float* rope = (const float*)d_in[1];
  const float* bq   = (const float*)d_in[3];
  const float* bk   = (const float*)d_in[5];
  const float* bv   = (const float*)d_in[7];
  const float* bo   = (const float*)d_in[9];
  const float* qn_w = (const float*)d_in[10];
  const float* kn_w = (const float*)d_in[11];
  const float* Wf[4] = { (const float*)d_in[2], (const float*)d_in[4],
                         (const float*)d_in[6], (const float*)d_in[8] };
  float* out = (float*)d_out;

  char* ws = (char*)d_ws;
  size_t off = 0;
  auto take = [&](size_t bytes) -> _Float16* {
    _Float16* p = (_Float16*)(ws + off);
    off += (bytes + 255) & ~(size_t)255;
    return p;
  };
  const size_t SC = (size_t)S_ * C_, CC = (size_t)C_ * C_;
  _Float16* xh  = take(SC * 2);
  _Float16* wh[4];
  for (int i = 0; i < 4; ++i) wh[i] = take(CC * 2);  // Wq,Wk,Wv,Wo (f16)
  _Float16* qraw = take(SC * 2);
  _Float16* kraw = take(SC * 2);
  _Float16* vraw = take(SC * 2);
  _Float16* vt   = take(SC * 2);
  _Float16* oh   = take(SC * 2);

  // 1) convert x and weights to f16 (L2-resident; trivial vs 180 GFLOP)
  f32_to_f16_kernel<<<(int)((SC + 255) / 256), 256, 0, stream>>>(x, xh, (int)SC);
  for (int i = 0; i < 4; ++i)
    f32_to_f16_kernel<<<(int)((CC + 255) / 256), 256, 0, stream>>>(Wf[i], wh[i], (int)CC);

  // 2) QKV projections via WMMA (32x64 per wave)
  const int gemm_blocks = (S_ / 32) * (C_ / 64) / 8;  // 8 waves per block
  wmma_gemm_xwt<_Float16><<<gemm_blocks, 256, 0, stream>>>(xh, wh[0], bq, qraw, S_, C_, C_);
  wmma_gemm_xwt<_Float16><<<gemm_blocks, 256, 0, stream>>>(xh, wh[1], bk, kraw, S_, C_, C_);
  wmma_gemm_xwt<_Float16><<<gemm_blocks, 256, 0, stream>>>(xh, wh[2], bv, vraw, S_, C_, C_);

  // 3) RMS-norm + RoPE on q,k (one wave per (s,h))
  norm_rope_kernel<<<(S_ * H_) / 8, 256, 0, stream>>>(qraw, kraw, rope, qn_w, kn_w);

  // 4) V -> per-head [D][S] for K-contiguous PV fragments
  transpose_v_kernel<<<(int)((SC + 255) / 256), 256, 0, stream>>>(vraw, vt);

  // 5) flash attention with TDM-staged K tiles (2 waves/block)
  flash_attn_kernel<<<(H_ * (S_ / 16)) / 2, 64, 0, stream>>>(qraw, kraw, vt, oh);

  // 6) output projection -> f32 d_out
  wmma_gemm_xwt<float><<<gemm_blocks, 256, 0, stream>>>(oh, wh[3], bo, out, S_, C_, C_);
}
